// EdgeScoringNetwork_37598143709240
// MI455X (gfx1250) — matrix operand order repaired and checked by reading, compiled, not run on hardware
//
#include <hip/hip_runtime.h>
#include <stdint.h>

#define BB 4
#define NNODE 4096
#define DEG 16
#define DD 128
#define EDIM 256
#define EE (NNODE * DEG)   // 65536 edges per batch
#define BN_EPS 1e-5f
#define NW 2               // waves per block in score kernel
#define MT 2               // 16-row M-subtiles per wave (32 edges/wave)

typedef __attribute__((ext_vector_type(16))) __bf16 v16bf;
typedef __attribute__((ext_vector_type(8)))  float  v8f;

union Frag { v16bf v; uint4 q[2]; };

__device__ __forceinline__ v8f wmma_bf16(v16bf a, v16bf b, v8f c) {
  // (neg_a, A, neg_b, B, c_mod, C, reuse_a, reuse_b)
  return __builtin_amdgcn_wmma_f32_16x16x32_bf16(false, a, false, b, (short)0, c, false, false);
}

// A fragment (16x32 bf16) from an LDS tile [32][stride], rows mbase..mbase+15, k-chunk `ktile`.
// lane L: row = mbase + (L&15) ; lanes 0-15 hold K = 0..7,16..23 ; lanes 16-31 hold K = 8..15,24..31
__device__ __forceinline__ v16bf ldsA(const __bf16* tile, int stride, int ktile, int lane, int mbase) {
  int row = mbase + (lane & 15);
  int kb  = (lane >> 4) * 8;
  int k0  = ktile * 32 + kb;
  Frag f;
  f.q[0] = *(const uint4*)(tile + row * stride + k0);
  f.q[1] = *(const uint4*)(tile + row * stride + k0 + 16);
  return f.v;
}

// B fragment (32x16 bf16) from a transposed weight WT[N][K] in global (bf16).
// lane L: col = L&15 ; lanes 0-15 hold K = 0..7,16..23 ; lanes 16-31 hold K = 8..15,24..31
__device__ __forceinline__ v16bf gB(const __bf16* WT, int K, int ntile, int ktile, int lane) {
  int col = lane & 15;
  int kb  = (lane >> 4) * 8;
  const __bf16* p = WT + (size_t)(ntile * 16 + col) * K + ktile * 32 + kb;
  Frag f;
  f.q[0] = *(const uint4*)(p);
  f.q[1] = *(const uint4*)(p + 16);
  return f.v;
}

// Store a 16x16 f32 C fragment as bf16 into LDS tile [32][stride], rows mbase..mbase+15, col tile `ntile`.
// VGPR r, lane L: row = mbase + r + (L>=16 ? 8 : 0), col = ntile*16 + (L&15)
__device__ __forceinline__ void storeC(__bf16* tile, int stride, int ntile, int lane, int mbase, v8f c) {
  int col = ntile * 16 + (lane & 15);
  int r0  = mbase + (lane >> 4) * 8;
#pragma unroll
  for (int r = 0; r < 8; r++) tile[(r0 + r) * stride + col] = (__bf16)c[r];
}

// ---------------- kernel 1: l2-normalize node features -> bf16 ----------------
__global__ void k_norm_nodes(const float* __restrict__ nf, __bf16* __restrict__ nfb) {
  int wave = (blockIdx.x * blockDim.x + threadIdx.x) >> 5;
  int lane = threadIdx.x & 31;
  if (wave >= BB * NNODE) return;
  const float4 x = *(const float4*)(nf + (size_t)wave * DD + lane * 4);
  float ss = x.x * x.x + x.y * x.y + x.z * x.z + x.w * x.w;
#pragma unroll
  for (int off = 16; off; off >>= 1) ss += __shfl_xor(ss, off, 32);
  float inv = 1.0f / fmaxf(sqrtf(ss), 1e-12f);
  __bf16 o[4];
  o[0] = (__bf16)(x.x * inv); o[1] = (__bf16)(x.y * inv);
  o[2] = (__bf16)(x.z * inv); o[3] = (__bf16)(x.w * inv);
  *(uint2*)(nfb + (size_t)wave * DD + lane * 4) = *(const uint2*)o;
}

// ---------------- kernel 2: transpose weights to bf16, fold BN ----------------
__global__ void k_prep(const float* __restrict__ Wa1, const float* __restrict__ Wa2,
                       const float* __restrict__ W1,  const float* __restrict__ W2,
                       const float* __restrict__ b1,  const float* __restrict__ g1,
                       const float* __restrict__ be1, const float* __restrict__ m1,
                       const float* __restrict__ v1,
                       const float* __restrict__ b2,  const float* __restrict__ g2,
                       const float* __restrict__ be2, const float* __restrict__ m2,
                       const float* __restrict__ v2,
                       __bf16* __restrict__ Wa1T, __bf16* __restrict__ Wa2T,
                       __bf16* __restrict__ W1T,  __bf16* __restrict__ W2T,
                       float* __restrict__ s1, float* __restrict__ t1,
                       float* __restrict__ s2, float* __restrict__ t2) {
  int i = blockIdx.x * blockDim.x + threadIdx.x;
  if (i < 128 * 256) { int n = i / 256, k = i % 256; Wa1T[i] = (__bf16)Wa1[k * 128 + n]; return; }
  i -= 128 * 256;
  if (i < 128 * 128) { int n = i / 128, k = i % 128; Wa2T[i] = (__bf16)Wa2[k * 128 + n]; return; }
  i -= 128 * 128;
  if (i < 256 * 384) { int n = i / 384, k = i % 384; W1T[i] = (__bf16)W1[k * 256 + n]; return; }
  i -= 256 * 384;
  if (i < 128 * 256) { int n = i / 256, k = i % 256; W2T[i] = (__bf16)W2[k * 128 + n]; return; }
  i -= 128 * 256;
  if (i < 256) { float s = g1[i] * rsqrtf(v1[i] + BN_EPS); s1[i] = s; t1[i] = (b1[i] - m1[i]) * s + be1[i]; return; }
  i -= 256;
  if (i < 128) { float s = g2[i] * rsqrtf(v2[i] + BN_EPS); s2[i] = s; t2[i] = (b2[i] - m2[i]) * s + be2[i]; return; }
}

// ---------------- kernel 3: WMMA edge-score pipeline (32 edges per wave) ----------------
__global__ __launch_bounds__(NW * 32)
void k_scores(const __bf16* __restrict__ nfb,
              const int* __restrict__ src_idx, const int* __restrict__ tgt_idx,
              const __bf16* __restrict__ Wa1T, const float* __restrict__ ba1,
              const __bf16* __restrict__ Wa2T, const float* __restrict__ ba2,
              const __bf16* __restrict__ W1T,  const float* __restrict__ s1, const float* __restrict__ t1,
              const __bf16* __restrict__ W2T,  const float* __restrict__ s2, const float* __restrict__ t2,
              const float* __restrict__ W3,    const float* __restrict__ b3,
              float* __restrict__ scores) {
  __shared__ __bf16 ef_s[NW][32][384];   // [sf | tf | |sf-tf|*att]
  __shared__ __bf16 h1_s[NW][32][256];   // layer-1 hidden
  __shared__ __bf16 h2_s[NW][32][128];   // attention hidden / layer-2 hidden (reused)

  const int lane = threadIdx.x & 31;
  const int wv   = threadIdx.x >> 5;
  const int gw   = blockIdx.x * NW + wv;     // 32-edge tile id, 0 .. B*E/32-1
  const int b    = gw >> 11;                 // EE/32 = 2048 tiles per batch
  const int t32  = gw & 2047;
  const int row  = lane & 15;
  const int half = lane >> 4;

  __bf16 (*ef)[384] = ef_s[wv];
  __bf16 (*h1)[256] = h1_s[wv];
  __bf16 (*h2)[128] = h2_s[wv];

  // ---- gather normalized endpoint features into ef[:, 0:256] (bf16), 2 rows per lane
#pragma unroll
  for (int m = 0; m < MT; m++) {
    const int e  = t32 * 32 + m * 16 + row;
    const int si = src_idx[b * EE + e];
    const int ti = tgt_idx[b * EE + e];
    const uint4* gs = (const uint4*)(nfb + ((size_t)(b * NNODE + si)) * DD + half * 64);
    const uint4* gt = (const uint4*)(nfb + ((size_t)(b * NNODE + ti)) * DD + half * 64);
    uint4* es = (uint4*)(&ef[m * 16 + row][half * 64]);
    uint4* et = (uint4*)(&ef[m * 16 + row][128 + half * 64]);
#pragma unroll
    for (int i = 0; i < 8; i++) es[i] = gs[i];
#pragma unroll
    for (int i = 0; i < 8; i++) et[i] = gt[i];
  }

  // ---- GEMM1: hid = relu(raw(32x256) @ Wa1T + ba1) -> h2 buffer (32x128)
#pragma unroll 1
  for (int nt = 0; nt < 8; nt++) {
    v8f a0 = {0, 0, 0, 0, 0, 0, 0, 0};
    v8f a1 = {0, 0, 0, 0, 0, 0, 0, 0};
#pragma unroll
    for (int kt = 0; kt < 8; kt++) {
      v16bf bf = gB(Wa1T, 256, nt, kt, lane);
      a0 = wmma_bf16(ldsA(&ef[0][0], 384, kt, lane, 0),  bf, a0);
      a1 = wmma_bf16(ldsA(&ef[0][0], 384, kt, lane, 16), bf, a1);
    }
    float bias = ba1[nt * 16 + (lane & 15)];
    v8f r0, r1;
#pragma unroll
    for (int i = 0; i < 8; i++) { r0[i] = fmaxf(a0[i] + bias, 0.0f); r1[i] = fmaxf(a1[i] + bias, 0.0f); }
    storeC(&h2[0][0], 128, nt, lane, 0,  r0);
    storeC(&h2[0][0], 128, nt, lane, 16, r1);
  }

  // ---- GEMM2: att_pre = hid(32x128) @ Wa2T + ba2 (kept in registers)
  v8f ap[MT][8];
#pragma unroll 1
  for (int nt = 0; nt < 8; nt++) {
    v8f a0 = {0, 0, 0, 0, 0, 0, 0, 0};
    v8f a1 = {0, 0, 0, 0, 0, 0, 0, 0};
#pragma unroll
    for (int kt = 0; kt < 4; kt++) {
      v16bf bf = gB(Wa2T, 128, nt, kt, lane);
      a0 = wmma_bf16(ldsA(&h2[0][0], 128, kt, lane, 0),  bf, a0);
      a1 = wmma_bf16(ldsA(&h2[0][0], 128, kt, lane, 16), bf, a1);
    }
    float bias = ba2[nt * 16 + (lane & 15)];
#pragma unroll
    for (int i = 0; i < 8; i++) { a0[i] += bias; a1[i] += bias; }
    ap[0][nt] = a0;
    ap[1][nt] = a1;
  }

  // ---- softmax over 128 cols per row (cols spread over 16-lane halves x 8 ntiles)
#pragma unroll
  for (int m = 0; m < MT; m++) {
    float mx[8], sm[8];
#pragma unroll
    for (int i = 0; i < 8; i++) {
      float mv = -3.4e38f;
#pragma unroll
      for (int nt = 0; nt < 8; nt++) mv = fmaxf(mv, ap[m][nt][i]);
      mx[i] = mv;
    }
#pragma unroll
    for (int i = 0; i < 8; i++)
#pragma unroll
      for (int off = 8; off; off >>= 1) mx[i] = fmaxf(mx[i], __shfl_xor(mx[i], off, 32));
#pragma unroll
    for (int i = 0; i < 8; i++) {
      float s = 0.0f;
#pragma unroll
      for (int nt = 0; nt < 8; nt++) { float ev = __expf(ap[m][nt][i] - mx[i]); ap[m][nt][i] = ev; s += ev; }
      sm[i] = s;
    }
#pragma unroll
    for (int i = 0; i < 8; i++)
#pragma unroll
      for (int off = 8; off; off >>= 1) sm[i] += __shfl_xor(sm[i], off, 32);
#pragma unroll
    for (int i = 0; i < 8; i++) {
      float inv = 1.0f / sm[i];
#pragma unroll
      for (int nt = 0; nt < 8; nt++) ap[m][nt][i] *= inv;
    }
  }

  // ---- ef[:, 256:384] = |sf - tf| * att
#pragma unroll 1
  for (int nt = 0; nt < 8; nt++) {
    int col = nt * 16 + (lane & 15);
#pragma unroll
    for (int m = 0; m < MT; m++) {
      int r0 = m * 16 + half * 8;
#pragma unroll
      for (int r = 0; r < 8; r++) {
        int rw   = r0 + r;
        float sf = (float)ef[rw][col];
        float tf = (float)ef[rw][128 + col];
        ef[rw][256 + col] = (__bf16)(fabsf(sf - tf) * ap[m][nt][r]);
      }
    }
  }

  // ---- GEMM3: h1 = relu(BNfold(ef(32x384) @ W1T))
#pragma unroll 1
  for (int nt = 0; nt < 16; nt++) {
    v8f a0 = {0, 0, 0, 0, 0, 0, 0, 0};
    v8f a1 = {0, 0, 0, 0, 0, 0, 0, 0};
#pragma unroll
    for (int kt = 0; kt < 12; kt++) {
      v16bf bf = gB(W1T, 384, nt, kt, lane);
      a0 = wmma_bf16(ldsA(&ef[0][0], 384, kt, lane, 0),  bf, a0);
      a1 = wmma_bf16(ldsA(&ef[0][0], 384, kt, lane, 16), bf, a1);
    }
    int col = nt * 16 + (lane & 15);
    float sc = s1[col], sh = t1[col];
    v8f r0, r1;
#pragma unroll
    for (int i = 0; i < 8; i++) {
      r0[i] = fmaxf(a0[i] * sc + sh, 0.0f);
      r1[i] = fmaxf(a1[i] * sc + sh, 0.0f);
    }
    storeC(&h1[0][0], 256, nt, lane, 0,  r0);
    storeC(&h1[0][0], 256, nt, lane, 16, r1);
  }

  // ---- GEMM4: h2 = relu(BNfold(h1(32x256) @ W2T))
#pragma unroll 1
  for (int nt = 0; nt < 8; nt++) {
    v8f a0 = {0, 0, 0, 0, 0, 0, 0, 0};
    v8f a1 = {0, 0, 0, 0, 0, 0, 0, 0};
#pragma unroll
    for (int kt = 0; kt < 8; kt++) {
      v16bf bf = gB(W2T, 256, nt, kt, lane);
      a0 = wmma_bf16(ldsA(&h1[0][0], 256, kt, lane, 0),  bf, a0);
      a1 = wmma_bf16(ldsA(&h1[0][0], 256, kt, lane, 16), bf, a1);
    }
    int col = nt * 16 + (lane & 15);
    float sc = s2[col], sh = t2[col];
    v8f r0, r1;
#pragma unroll
    for (int i = 0; i < 8; i++) {
      r0[i] = fmaxf(a0[i] * sc + sh, 0.0f);
      r1[i] = fmaxf(a1[i] * sc + sh, 0.0f);
    }
    storeC(&h2[0][0], 128, nt, lane, 0,  r0);
    storeC(&h2[0][0], 128, nt, lane, 16, r1);
  }

  // ---- logits = h2 @ W3 + b3 ; score = sigmoid
#pragma unroll
  for (int m = 0; m < MT; m++) {
    float acc = 0.0f;
#pragma unroll
    for (int j = 0; j < 64; j++) acc += (float)h2[m * 16 + row][half * 64 + j] * W3[half * 64 + j];
    acc += __shfl_xor(acc, 16, 32);
    float score = 1.0f / (1.0f + __expf(-(acc + b3[0])));
    if (lane < 16) scores[(size_t)b * EE + t32 * 32 + m * 16 + row] = score;
  }
}

// ---------------- kernel 4: exact 50th-percentile via byte radix-select ----------------
__global__ void k_threshold(const float* __restrict__ scores, float* __restrict__ thr) {
  __shared__ unsigned hist[256];
  __shared__ unsigned s_prefix, s_k;
  const int b = blockIdx.x;
  const unsigned* u = (const unsigned*)(scores + (size_t)b * EE);
  if (threadIdx.x == 0) { s_prefix = 0u; s_k = 32767u; }  // kth-smallest, k=E-1-thr_idx
  __syncthreads();
  for (int pass = 0; pass < 4; pass++) {
    int shift = 24 - pass * 8;
    for (int i = threadIdx.x; i < 256; i += blockDim.x) hist[i] = 0;
    __syncthreads();
    unsigned prefix = s_prefix;
    unsigned pmask  = (pass == 0) ? 0u : (0xFFFFFFFFu << (shift + 8));
    for (int i = threadIdx.x; i < EE; i += blockDim.x) {
      unsigned v = u[i];
      if ((v & pmask) == prefix) atomicAdd(&hist[(v >> shift) & 255u], 1u);
    }
    __syncthreads();
    if (threadIdx.x == 0) {
      unsigned k = s_k, d = 0;
      for (; d < 255; d++) { if (k < hist[d]) break; k -= hist[d]; }
      s_prefix = prefix | (d << shift);
      s_k = k;
    }
    __syncthreads();
  }
  if (threadIdx.x == 0) thr[b] = __uint_as_float(s_prefix);
}

// ---------------- kernel 5: zero the dense output ----------------
__global__ void k_zero(float4* __restrict__ out, size_t n4) {
  size_t i = (size_t)blockIdx.x * blockDim.x + threadIdx.x;
  if (i < n4) out[i] = make_float4(0.f, 0.f, 0.f, 0.f);
}

// ---------------- kernel 6: per-source repair + scatter ----------------
__global__ void k_scatter(const float* __restrict__ scores, const float* __restrict__ thr,
                          const int* __restrict__ src_idx, const int* __restrict__ tgt_idx,
                          const int* __restrict__ minep, float* __restrict__ out) {
  int g = blockIdx.x * blockDim.x + threadIdx.x;  // 0 .. B*N-1 (one source group)
  if (g >= BB * NNODE) return;
  int b = g >> 12;
  int base = g * DEG;  // == b*EE + node*DEG
  float s[DEG];
#pragma unroll
  for (int i = 0; i < DEG; i++) s[i] = scores[(size_t)base + i];
  float t = thr[b];
  int me = minep[0];
  int active = 0;
#pragma unroll
  for (int i = 0; i < DEG; i++) active += (s[i] >= t) ? 1 : 0;
  int need = 0;
  if (active < me) { need = me - active; if (need > DEG) need = DEG; }
#pragma unroll 1
  for (int i = 0; i < DEG; i++) {
    int rank = 0;  // stable descending rank (ties broken by index, matches argsort(argsort(-s)))
#pragma unroll
    for (int j = 0; j < DEG; j++)
      if (s[j] > s[i] || (s[j] == s[i] && j < i)) rank++;
    bool keep = (s[i] >= t) || (rank < need);
    float val = keep ? s[i] : 0.0f;
    int eidx = base + i;
    int src = src_idx[eidx];
    int tgt = tgt_idx[eidx];
    out[((size_t)b * NNODE + src) * NNODE + tgt] = val;
  }
}

extern "C" void kernel_launch(void* const* d_in, const int* in_sizes, int n_in,
                              void* d_out, int out_size, void* d_ws, size_t ws_size,
                              hipStream_t stream) {
  const float* node_feat = (const float*)d_in[0];
  const int*   src_idx   = (const int*)d_in[1];
  const int*   tgt_idx   = (const int*)d_in[2];
  const float* Wa1 = (const float*)d_in[3];
  const float* ba1 = (const float*)d_in[4];
  const float* Wa2 = (const float*)d_in[5];
  const float* ba2 = (const float*)d_in[6];
  const float* W1  = (const float*)d_in[7];
  const float* b1  = (const float*)d_in[8];
  const float* g1  = (const float*)d_in[9];
  const float* be1 = (const float*)d_in[10];
  const float* m1  = (const float*)d_in[11];
  const float* v1  = (const float*)d_in[12];
  const float* W2  = (const float*)d_in[13];
  const float* b2  = (const float*)d_in[14];
  const float* g2  = (const float*)d_in[15];
  const float* be2 = (const float*)d_in[16];
  const float* m2  = (const float*)d_in[17];
  const float* v2  = (const float*)d_in[18];
  const float* W3  = (const float*)d_in[19];
  const float* b3  = (const float*)d_in[20];
  const int*   minep = (const int*)d_in[21];

  char* ws = (char*)d_ws;
  size_t off = 0;
  auto alloc = [&](size_t bytes) -> void* {
    off = (off + 255) & ~(size_t)255;
    void* p = ws + off;
    off += bytes;
    return p;
  };
  __bf16* nfb  = (__bf16*)alloc((size_t)BB * NNODE * DD * 2);
  __bf16* Wa1T = (__bf16*)alloc(128 * 256 * 2);
  __bf16* Wa2T = (__bf16*)alloc(128 * 128 * 2);
  __bf16* W1T  = (__bf16*)alloc(256 * 384 * 2);
  __bf16* W2T  = (__bf16*)alloc(128 * 256 * 2);
  float* s1 = (float*)alloc(256 * 4);
  float* t1 = (float*)alloc(256 * 4);
  float* s2 = (float*)alloc(128 * 4);
  float* t2 = (float*)alloc(128 * 4);
  float* scores = (float*)alloc((size_t)BB * EE * 4);
  float* thr    = (float*)alloc(BB * 4);

  k_norm_nodes<<<BB * NNODE / 8, 256, 0, stream>>>(node_feat, nfb);

  int prep_total = 128 * 256 + 128 * 128 + 256 * 384 + 128 * 256 + 256 + 128;
  k_prep<<<(prep_total + 255) / 256, 256, 0, stream>>>(
      Wa1, Wa2, W1, W2, b1, g1, be1, m1, v1, b2, g2, be2, m2, v2,
      Wa1T, Wa2T, W1T, W2T, s1, t1, s2, t2);

  k_scores<<<(BB * EE / 32) / NW, NW * 32, 0, stream>>>(
      nfb, src_idx, tgt_idx, Wa1T, ba1, Wa2T, ba2,
      W1T, s1, t1, W2T, s2, t2, W3, b3, scores);

  k_threshold<<<BB, 256, 0, stream>>>(scores, thr);

  size_t n4 = (size_t)BB * NNODE * NNODE / 4;
  k_zero<<<(unsigned)(n4 / 256), 256, 0, stream>>>((float4*)d_out, n4);

  k_scatter<<<(BB * NNODE + 255) / 256, 256, 0, stream>>>(
      scores, thr, src_idx, tgt_idx, minep, (float*)d_out);
}
